// MambaBlock_76338748719399
// MI455X (gfx1250) — matrix operand order repaired and checked by reading, not compiled, not run on hardware
//
#include <hip/hip_runtime.h>
#include <hip/hip_bf16.h>

// ---------------------------------------------------------------------------
// Mamba block for MI455X (gfx1250, wave32, WMMA, async-LDS staging).
// GEMMs: v_wmma_f32_16x16x32_bf16; B tiles staged to LDS with
// global_load_async_to_lds_b128 (double-buffered), A streamed as bf16 b128.
// Scan : lane = (channel, state) layout, v_exp_f32 + shfl_xor reduction.
// ---------------------------------------------------------------------------

#define D_MODEL  1024
#define D_STATE  16
#define D_CONV   4
#define D_INNER  2048
#define DT_RANK  64
#define BATCH    2
#define SEQ      2048
#define MROWS    (BATCH * SEQ)           // 4096 token rows
#define N_IN2    (2 * D_INNER)           // 4096
#define N_DBC    (DT_RANK + 2 * D_STATE) // 96
#define N_DBC_PAD 128                    // padded row count for XPt staging

#define LDS_ROW  40                      // bf16 elements per LDS B row (80 B)

typedef __bf16 bf16_t;
typedef __attribute__((ext_vector_type(16))) __bf16 v16bf;
typedef __attribute__((ext_vector_type(8)))  __bf16 v8bf;
typedef __attribute__((ext_vector_type(8)))  float  v8f;
typedef int vsi4 __attribute__((vector_size(16)));   // matches builtin param type

#if defined(__has_builtin)
#if __has_builtin(__builtin_amdgcn_global_load_async_to_lds_b128) && \
    __has_builtin(__builtin_amdgcn_s_wait_asynccnt)
#define USE_ASYNC_LDS 1
#endif
#endif

#if defined(USE_ASYNC_LDS)
#define WAIT_ASYNC(n) __builtin_amdgcn_s_wait_asynccnt(n)
#else
#define WAIT_ASYNC(n) ((void)0)
#endif

static __device__ __forceinline__ bf16_t f2bf(float f) {
    unsigned u = __builtin_bit_cast(unsigned, f);
    u += 0x7fffu + ((u >> 16) & 1u);              // round-to-nearest-even
    unsigned short h = (unsigned short)(u >> 16);
    return __builtin_bit_cast(bf16_t, h);
}

static __device__ __forceinline__ float silu(float x) {
    return x / (1.0f + __expf(-x));
}

// 16-byte global -> LDS copy (async when available)
static __device__ __forceinline__ void copy16_g2l(const bf16_t* src, bf16_t* dst) {
#if defined(USE_ASYNC_LDS)
    __builtin_amdgcn_global_load_async_to_lds_b128(
        (vsi4*)(void*)src,
        (__attribute__((address_space(3))) vsi4*)(void*)dst, 0, 0);
#else
    *reinterpret_cast<uint4*>(dst) = *reinterpret_cast<const uint4*>(src);
#endif
}

// ---------------------------------------------------------------------------
// Transpose + fp32->bf16 cast:  in (rows x cols) -> out (cols x rows)
// ---------------------------------------------------------------------------
__global__ void tcast_kernel(const float* __restrict__ in, bf16_t* __restrict__ out,
                             int rows, int cols) {
    __shared__ float tile[32][33];
    int c0 = blockIdx.x * 32, r0 = blockIdx.y * 32;
    int tx = threadIdx.x, ty = threadIdx.y;      // block (32, 8)
    #pragma unroll
    for (int i = ty; i < 32; i += 8) {
        int r = r0 + i, c = c0 + tx;
        tile[i][tx] = (r < rows && c < cols) ? in[(size_t)r * cols + c] : 0.0f;
    }
    __syncthreads();
    #pragma unroll
    for (int i = ty; i < 32; i += 8) {
        int r = r0 + tx, c = c0 + i;
        if (r < rows && c < cols)
            out[(size_t)c * rows + r] = f2bf(tile[tx][i]);
    }
}

// Contiguous fp32 -> bf16 cast (4 elements per thread)
__global__ void cast_bf16_kernel(const float* __restrict__ in, bf16_t* __restrict__ out,
                                 size_t n4) {
    size_t i = (size_t)blockIdx.x * blockDim.x + threadIdx.x;
    if (i >= n4) return;
    float4 v = reinterpret_cast<const float4*>(in)[i];
    bf16_t o[4] = {f2bf(v.x), f2bf(v.y), f2bf(v.z), f2bf(v.w)};
    reinterpret_cast<uint2*>(out)[i] = *reinterpret_cast<uint2*>(o);
}

// Strided fp32 -> packed bf16 cast (dt slice of dbc: stride 96 -> 64)
__global__ void cast_strided_kernel(const float* __restrict__ in, long ld_in,
                                    bf16_t* __restrict__ out, int rows, int cols) {
    size_t i = (size_t)blockIdx.x * blockDim.x + threadIdx.x;
    if (i >= (size_t)rows * cols) return;
    size_t r = i / cols;
    int c = (int)(i % cols);
    out[i] = f2bf(in[r * ld_in + c]);
}

// ---------------------------------------------------------------------------
// WMMA GEMM:  C(M x N) = A(M x K, bf16, row stride lda) * Bt(N x K, bf16)^T
// Block = 128 threads = 4 waves; block tile 64m x 64n; wave = 16m x 64n.
// B tile (64n x 32k) staged to LDS, double-buffered, async when available.
// EPI: 0 = plain store, 1 = bias + softplus (dt projection)
// ---------------------------------------------------------------------------
template <int EPI>
__global__ void gemm_wmma_kernel(const bf16_t* __restrict__ A, long lda,
                                 const bf16_t* __restrict__ Bt,
                                 float* __restrict__ C, long ldc,
                                 const float* __restrict__ bias,
                                 int M, int N, int K) {
    __shared__ bf16_t Blds[2][64 * LDS_ROW];     // 2 x 5120 B

    const int lane = threadIdx.x & 31;
    const int wave = threadIdx.x >> 5;
    const int m0   = (blockIdx.y * 4 + wave) * 16;
    const int n0   = blockIdx.x * 64;
    const int nlo  = lane & 15;
    const int kh   = lane >> 4;                  // 0/1 half-wave

    // staging geometry: thread -> (row, 32-byte half)
    const int sr = threadIdx.x >> 1;             // 0..63 (B tile row)
    const int sh = (threadIdx.x & 1) * 16;       // 0 or 16 bf16 (32 B half)

    v8f acc[4];
    #pragma unroll
    for (int t = 0; t < 4; ++t)
        #pragma unroll
        for (int r = 0; r < 8; ++r) acc[t][r] = 0.0f;

    const bf16_t* arow = A + (size_t)(m0 + nlo) * lda;

    // ---- stage first B tile ----
    {
        const bf16_t* src = Bt + (size_t)(n0 + sr) * K + sh;
        bf16_t* dst = &Blds[0][sr * LDS_ROW + sh];
        copy16_g2l(src, dst);
        copy16_g2l(src + 8, dst + 8);
    }

    int buf = 0;
    for (int k0 = 0; k0 < K; k0 += 32) {
        // A fragment for this k-step (global bf16, two b128 per lane)
        v8bf a0 = *reinterpret_cast<const v8bf*>(arow + k0 + kh * 16);
        v8bf a1 = *reinterpret_cast<const v8bf*>(arow + k0 + kh * 16 + 8);
        v16bf afrag;
        #pragma unroll
        for (int i = 0; i < 8; ++i) { afrag[i] = a0[i]; afrag[8 + i] = a1[i]; }

        __syncthreads();                         // prev-iter reads of next buf done
        const int cur = buf;
        if (k0 + 32 < K) {                       // stage next tile into other buffer
            buf ^= 1;
            const bf16_t* src = Bt + (size_t)(n0 + sr) * K + (k0 + 32) + sh;
            bf16_t* dst = &Blds[buf][sr * LDS_ROW + sh];
            copy16_g2l(src, dst);
            copy16_g2l(src + 8, dst + 8);
            WAIT_ASYNC(2);                       // current buffer's loads retired
        } else {
            WAIT_ASYNC(0);
        }
        __syncthreads();                         // whole current tile visible

        #pragma unroll
        for (int t = 0; t < 4; ++t) {
            const bf16_t* lp = &Blds[cur][(t * 16 + nlo) * LDS_ROW + kh * 16];
            v8bf b0 = *reinterpret_cast<const v8bf*>(lp);
            v8bf b1 = *reinterpret_cast<const v8bf*>(lp + 8);
            v16bf bfrag;
            #pragma unroll
            for (int i = 0; i < 8; ++i) { bfrag[i] = b0[i]; bfrag[8 + i] = b1[i]; }
            acc[t] = __builtin_amdgcn_wmma_f32_16x16x32_bf16(
                false, afrag, false, bfrag, (short)0, acc[t], false, false);
        }
    }

    // ---- store D (f32 16x16 layout: lane = col n, rows kh*8 + r) ----
    #pragma unroll
    for (int t = 0; t < 4; ++t) {
        int n = n0 + t * 16 + nlo;
        if (n >= N) continue;
        #pragma unroll
        for (int r = 0; r < 8; ++r) {
            int m = m0 + kh * 8 + r;
            float v = acc[t][r];
            if (EPI == 1) {                      // bias + softplus
                v += bias[n];
                v = (v > 30.0f) ? v : __logf(1.0f + __expf(v));
            }
            C[(size_t)m * ldc + n] = v;
        }
    }
}

// ---------------------------------------------------------------------------
// Causal depthwise conv1d (width 4) + bias + SiLU. Writes fp32 + bf16 copies.
// ---------------------------------------------------------------------------
__global__ void conv_silu_kernel(const float* __restrict__ xr,
                                 const float* __restrict__ conv_w,
                                 const float* __restrict__ conv_b,
                                 float* __restrict__ u,
                                 bf16_t* __restrict__ u_bf) {
    size_t idx = (size_t)blockIdx.x * blockDim.x + threadIdx.x;
    if (idx >= (size_t)MROWS * D_INNER) return;
    int d = (int)(idx % D_INNER);
    int l = (int)((idx / D_INNER) % SEQ);
    int b = (int)(idx / ((size_t)D_INNER * SEQ));
    float acc = conv_b[d];
    #pragma unroll
    for (int j = 0; j < D_CONV; ++j) {
        int ls = l - (D_CONV - 1) + j;
        if (ls >= 0)
            acc += conv_w[d * D_CONV + j] *
                   xr[(size_t)(b * SEQ + ls) * N_IN2 + d];
    }
    float s = silu(acc);
    size_t o = (size_t)(b * SEQ + l) * D_INNER + d;
    u[o] = s;
    u_bf[o] = f2bf(s);
}

// ---------------------------------------------------------------------------
// Selective scan. wave32 = 2 channels x 16 states.  lane = ch*16 + n.
// ---------------------------------------------------------------------------
__global__ void scan_kernel(const float* __restrict__ delta,
                            const float* __restrict__ u,
                            const float* __restrict__ dbc,
                            const float* __restrict__ A_log,
                            const float* __restrict__ Dvec,
                            float* __restrict__ yssm) {
    int lane = threadIdx.x & 31;
    int wavg = (int)((blockIdx.x * blockDim.x + threadIdx.x) >> 5);
    int chan = wavg * 2 + (lane >> 4);           // [0, BATCH*D_INNER)
    int b  = chan / D_INNER;
    int dd = chan % D_INNER;
    int n  = lane & 15;

    float Aval = -__expf(A_log[dd * D_STATE + n]);
    float Dd   = Dvec[dd];
    float h    = 0.0f;

    for (int t = 0; t < SEQ; ++t) {
        size_t r  = (size_t)b * SEQ + t;
        float dt  = delta[r * D_INNER + dd];
        float uu  = u[r * D_INNER + dd];
        float Bv  = dbc[r * N_DBC + DT_RANK + n];
        float Cv  = dbc[r * N_DBC + DT_RANK + D_STATE + n];
        float dA  = __expf(dt * Aval);
        h = fmaf(h, dA, dt * uu * Bv);
        float c = h * Cv;
        c += __shfl_xor(c, 1);
        c += __shfl_xor(c, 2);
        c += __shfl_xor(c, 4);
        c += __shfl_xor(c, 8);                   // reduce within 16-lane group
        if (n == 0) yssm[r * D_INNER + dd] = fmaf(Dd, uu, c);
    }
}

// ---------------------------------------------------------------------------
// Gating: ygate_bf = bf16( yssm * silu(res) ),  res = xr[:, 2048 + d]
// ---------------------------------------------------------------------------
__global__ void gate_kernel(const float* __restrict__ yssm,
                            const float* __restrict__ xr,
                            bf16_t* __restrict__ ygate_bf) {
    size_t idx = (size_t)blockIdx.x * blockDim.x + threadIdx.x;
    if (idx >= (size_t)MROWS * D_INNER) return;
    size_t row = idx / D_INNER;
    int d = (int)(idx % D_INNER);
    float res = xr[row * N_IN2 + D_INNER + d];
    ygate_bf[idx] = f2bf(yssm[idx] * silu(res));
}

// ---------------------------------------------------------------------------
extern "C" void kernel_launch(void* const* d_in, const int* in_sizes, int n_in,
                              void* d_out, int out_size, void* d_ws, size_t ws_size,
                              hipStream_t stream) {
    (void)in_sizes; (void)n_in; (void)out_size; (void)ws_size;

    const float* x         = (const float*)d_in[0];  // (B,L,1024)
    const float* in_proj_w = (const float*)d_in[1];  // (1024,4096)
    const float* conv_w    = (const float*)d_in[2];  // (2048,1,4)
    const float* conv_b    = (const float*)d_in[3];  // (2048)
    const float* x_proj_w  = (const float*)d_in[4];  // (2048,96)
    const float* dt_proj_w = (const float*)d_in[5];  // (64,2048)
    const float* dt_proj_b = (const float*)d_in[6];  // (2048)
    const float* A_log     = (const float*)d_in[7];  // (2048,16)
    const float* Dvec      = (const float*)d_in[8];  // (2048)
    const float* out_proj_w= (const float*)d_in[9];  // (2048,1024)

    // workspace carve-out (256B aligned)
    char* p = (char*)d_ws;
    auto carve = [&](size_t bytes) -> void* {
        void* r = (void*)p;
        p += (bytes + 255) & ~(size_t)255;
        return r;
    };
    bf16_t* W1t   = (bf16_t*)carve((size_t)N_IN2 * D_MODEL * 2);       // (4096,1024)
    bf16_t* XPt   = (bf16_t*)carve((size_t)N_DBC_PAD * D_INNER * 2);   // (128,2048) padded
    bf16_t* DTt   = (bf16_t*)carve((size_t)D_INNER * DT_RANK * 2);     // (2048,64)
    bf16_t* OPt   = (bf16_t*)carve((size_t)D_MODEL * D_INNER * 2);     // (1024,2048)
    bf16_t* x_bf  = (bf16_t*)carve((size_t)MROWS * D_MODEL * 2);       // (4096,1024)
    bf16_t* u_bf  = (bf16_t*)carve((size_t)MROWS * D_INNER * 2);       // (4096,2048)
    bf16_t* dt_bf = (bf16_t*)carve((size_t)MROWS * DT_RANK * 2);       // (4096,64)
    bf16_t* yg_bf = (bf16_t*)carve((size_t)MROWS * D_INNER * 2);       // (4096,2048)
    float*  xr    = (float*) carve((size_t)MROWS * N_IN2 * 4);         // (4096,4096)
    float*  u     = (float*) carve((size_t)MROWS * D_INNER * 4);
    float*  dbc   = (float*) carve((size_t)MROWS * N_DBC * 4);
    float*  delta = (float*) carve((size_t)MROWS * D_INNER * 4);
    float*  yssm  = (float*) carve((size_t)MROWS * D_INNER * 4);

    dim3 tb(32, 8);
    // weight transpose + bf16 cast
    tcast_kernel<<<dim3(N_IN2 / 32, D_MODEL / 32), tb, 0, stream>>>(in_proj_w, W1t, D_MODEL, N_IN2);
    tcast_kernel<<<dim3((N_DBC + 31) / 32, D_INNER / 32), tb, 0, stream>>>(x_proj_w, XPt, D_INNER, N_DBC);
    tcast_kernel<<<dim3(D_INNER / 32, DT_RANK / 32), tb, 0, stream>>>(dt_proj_w, DTt, DT_RANK, D_INNER);
    tcast_kernel<<<dim3(D_MODEL / 32, D_INNER / 32), tb, 0, stream>>>(out_proj_w, OPt, D_INNER, D_MODEL);

    // activation cast: x -> bf16
    {
        size_t n4 = (size_t)MROWS * D_MODEL / 4;
        cast_bf16_kernel<<<(unsigned)((n4 + 255) / 256), 256, 0, stream>>>(x, x_bf, n4);
    }

    // 1) xr = x @ in_proj_w                 (4096 x 4096, K=1024)
    gemm_wmma_kernel<0><<<dim3(N_IN2 / 64, MROWS / 64), 128, 0, stream>>>(
        x_bf, D_MODEL, W1t, xr, N_IN2, nullptr, MROWS, N_IN2, D_MODEL);

    // 2) u = silu(causal depthwise conv(xb) + b)   (fp32 + bf16)
    {
        size_t tot = (size_t)MROWS * D_INNER;
        conv_silu_kernel<<<(unsigned)((tot + 255) / 256), 256, 0, stream>>>(xr, conv_w, conv_b, u, u_bf);
    }

    // 3) dbc = u @ x_proj_w                 (4096 x 96, K=2048)
    gemm_wmma_kernel<0><<<dim3((N_DBC + 63) / 64, MROWS / 64), 128, 0, stream>>>(
        u_bf, D_INNER, XPt, dbc, N_DBC, nullptr, MROWS, N_DBC, D_INNER);

    // 3b) dt slice -> packed bf16
    {
        size_t tot = (size_t)MROWS * DT_RANK;
        cast_strided_kernel<<<(unsigned)((tot + 255) / 256), 256, 0, stream>>>(dbc, N_DBC, dt_bf, MROWS, DT_RANK);
    }

    // 4) delta = softplus(dt @ dt_proj_w + b)   (4096 x 2048, K=64)
    gemm_wmma_kernel<1><<<dim3(D_INNER / 64, MROWS / 64), 128, 0, stream>>>(
        dt_bf, DT_RANK, DTt, delta, D_INNER, dt_proj_b, MROWS, D_INNER, DT_RANK);

    // 5) selective scan (2048 waves, 2 channels x 16 states each)
    scan_kernel<<<(BATCH * D_INNER / 2) / 4, 128, 0, stream>>>(delta, u, dbc, A_log, Dvec, yssm);

    // 6) gating -> bf16
    {
        size_t tot = (size_t)MROWS * D_INNER;
        gate_kernel<<<(unsigned)((tot + 255) / 256), 256, 0, stream>>>(yssm, xr, yg_bf);
    }

    // 7) out = ygate @ out_proj_w           (4096 x 1024, K=2048)
    gemm_wmma_kernel<0><<<dim3(D_MODEL / 64, MROWS / 64), 128, 0, stream>>>(
        yg_bf, D_INNER, OPt, (float*)d_out, D_MODEL, nullptr, MROWS, D_MODEL, D_INNER);
}
